// GGNNEncoder_22325240004851
// MI455X (gfx1250) — compile-verified, matrix operand order checked by reading, not run on hardware
//
#include <hip/hip_runtime.h>
#include <hip/hip_bf16.h>

// ---------------------------------------------------------------------------
// GGNN encoder for MI455X (gfx1250): bf16 internals + v_wmma_f32_16x16x32_bf16
// + global_atomic_pk_add_bf16 scatter (L2-resident aggregation target).
// ---------------------------------------------------------------------------

typedef __bf16 bf16_t;
typedef __attribute__((ext_vector_type(16))) __bf16 v16bf;
typedef __attribute__((ext_vector_type(8)))  float  v8f;
typedef unsigned int u32;

// Load one WMMA 16-bit operand fragment for this lane.
// ISA layout (16-bit A/B, 16x32 / 32x16): lane holds two contiguous 8-element
// chunks: K=[kb, kb+7] and K=[kb+16, kb+23], where kb = k0 + 8*(lane>>4).
__device__ __forceinline__ v16bf load_frag(const bf16_t* p) {
  union { v16bf v; uint4 u[2]; } t;
  t.u[0] = *(const uint4*)(p);
  t.u[1] = *(const uint4*)(p + 16);
  return t.v;
}

__device__ __forceinline__ v8f wmma_bf16(v16bf a, v16bf b, v8f c) {
  return __builtin_amdgcn_wmma_f32_16x16x32_bf16(
      /*neg_a=*/false, a, /*neg_b=*/false, b,
      /*c_mod=*/(short)0, c, /*reuse_a=*/false, /*reuse_b=*/false);
}

// ---------------------------------------------------------------------------
__global__ void cvt_f32_bf16(const float* __restrict__ src,
                             bf16_t* __restrict__ dst, int n) {
  int i = blockIdx.x * blockDim.x + threadIdx.x;
  if (i < n) dst[i] = (bf16_t)src[i];
}

__global__ void zero_u32(u32* __restrict__ p, int n) {
  int i = blockIdx.x * blockDim.x + threadIdx.x;
  if (i < n) p[i] = 0u;
}

// ---------------------------------------------------------------------------
// Generic K=128, Nout=128 GEMM:  C = A(bf16 [M,128]) * B^T(bf16 [128,128]) + bias
// grid.y batches over B/C (edge types). One wave -> one 16-row x 128-col strip.
template <bool RELU, typename OutT>
__global__ void gemm128(const bf16_t* __restrict__ A,
                        const bf16_t* __restrict__ B0, long bStrideY,
                        OutT* __restrict__ C0, long cStrideY,
                        const float* __restrict__ bias, int Mtiles) {
  const int lane = threadIdx.x & 31;
  const int wave = threadIdx.x >> 5;
  const int rowtile = blockIdx.x * 8 + wave;
  if (rowtile >= Mtiles) return;                 // wave-uniform
  const bf16_t* B = B0 + (long)blockIdx.y * bStrideY;
  OutT* C = C0 + (long)blockIdx.y * cStrideY;

  const int m0  = rowtile * 16;
  const int rA  = m0 + (lane & 15);
  const int ksel = (lane >> 4) << 3;             // 0 or 8
  const int nlo = lane & 15;

  v8f acc[8];
#pragma unroll
  for (int n = 0; n < 8; ++n)
#pragma unroll
    for (int i = 0; i < 8; ++i) acc[n][i] = 0.0f;

#pragma unroll
  for (int kk = 0; kk < 128; kk += 32) {
    const int kb = kk + ksel;
    v16bf a = load_frag(A + (long)rA * 128 + kb);
#pragma unroll
    for (int n = 0; n < 8; ++n) {
      v16bf b = load_frag(B + (long)(n * 16 + nlo) * 128 + kb);
      acc[n] = wmma_bf16(a, b, acc[n]);
    }
  }

  const int mhi = (lane >> 4) * 8;               // C layout: M = v + 8*(lane>>4)
#pragma unroll
  for (int n = 0; n < 8; ++n) {
    const int col = n * 16 + nlo;
    const float bv = bias ? bias[col] : 0.0f;
#pragma unroll
    for (int v = 0; v < 8; ++v) {
      const int row = m0 + mhi + v;
      float x = acc[n][v] + bv;
      if (RELU) x = fmaxf(x, 0.0f);
      C[(long)row * 128 + col] = (OutT)x;
    }
  }
}

// ---------------------------------------------------------------------------
// Per-edge gather of trans[type][src] (128 bf16) and packed-bf16 atomic
// scatter-add into agg[dst]. agg is 25.6 MB -> L2-resident atomics.
__global__ void scatter_edges(const bf16_t* __restrict__ trans,
                              const int* __restrict__ esrc,
                              const int* __restrict__ edst,
                              const int* __restrict__ etype,
                              bf16_t* __restrict__ agg,
                              int nEdges, int nNodes) {
  const int lane = threadIdx.x & 31;
  const int wave = threadIdx.x >> 5;
  const int e = blockIdx.x * 8 + wave;
  if (e >= nEdges) return;                       // wave-uniform
  const int s = esrc[e], d = edst[e], t = etype[e];
  const uint2* srow = (const uint2*)(trans + ((long)t * nNodes + s) * 128);
  uint2 m = srow[lane];                          // 4 bf16 per lane
  bf16_t* daddr = agg + (long)d * 128 + lane * 4;
  asm volatile("global_atomic_pk_add_bf16 %0, %1, off"
               :: "v"(daddr), "v"(m.x) : "memory");
  asm volatile("global_atomic_pk_add_bf16 %0, %1, off offset:4"
               :: "v"(daddr), "v"(m.y) : "memory");
}

// ---------------------------------------------------------------------------
// Fused GRU cell + ReLU.  One wave -> one 16x16 tile of h'.
// gi = agg*wih^T, gh = h*whh^T (3 gates each, WMMA), then gate math.
__global__ void gru_fused(const bf16_t* __restrict__ agg,
                          const bf16_t* __restrict__ h,
                          const bf16_t* __restrict__ wih,  // [384,128]
                          const bf16_t* __restrict__ whh,  // [384,128]
                          const float* __restrict__ bih,   // [384]
                          const float* __restrict__ bhh,   // [384]
                          bf16_t* __restrict__ hout, int Mtiles) {
  const int lane = threadIdx.x & 31;
  const int wave = threadIdx.x >> 5;
  const int task = blockIdx.x * 8 + wave;        // rowtile*8 + coltile
  const int rowtile = task >> 3;
  if (rowtile >= Mtiles) return;                 // wave-uniform
  const int c0 = (task & 7) * 16;

  const int m0  = rowtile * 16;
  const int rA  = m0 + (lane & 15);
  const int ksel = (lane >> 4) << 3;
  const int nlo = lane & 15;

  v8f gi[3], gh[3];
#pragma unroll
  for (int g = 0; g < 3; ++g)
#pragma unroll
    for (int i = 0; i < 8; ++i) { gi[g][i] = 0.0f; gh[g][i] = 0.0f; }

#pragma unroll
  for (int kk = 0; kk < 128; kk += 32) {
    const int kb = kk + ksel;
    v16bf aA = load_frag(agg + (long)rA * 128 + kb);
    v16bf aH = load_frag(h   + (long)rA * 128 + kb);
#pragma unroll
    for (int g = 0; g < 3; ++g) {
      const long rB = (long)(g * 128 + c0 + nlo) * 128 + kb;
      v16bf bI = load_frag(wih + rB);
      v16bf bH = load_frag(whh + rB);
      gi[g] = wmma_bf16(aA, bI, gi[g]);
      gh[g] = wmma_bf16(aH, bH, gh[g]);
    }
  }

  const int col = c0 + nlo;
  const float br_i = bih[col],       br_h = bhh[col];
  const float bz_i = bih[128 + col], bz_h = bhh[128 + col];
  const float bn_i = bih[256 + col], bn_h = bhh[256 + col];
  const int mhi = (lane >> 4) * 8;
#pragma unroll
  for (int v = 0; v < 8; ++v) {
    const int row = m0 + mhi + v;
    float r = 1.0f / (1.0f + __expf(-(gi[0][v] + br_i + gh[0][v] + br_h)));
    float z = 1.0f / (1.0f + __expf(-(gi[1][v] + bz_i + gh[1][v] + bz_h)));
    float nn = tanhf(gi[2][v] + bn_i + r * (gh[2][v] + bn_h));
    float hv = (float)h[(long)row * 128 + col];
    float hn = (1.0f - z) * nn + z * hv;
    hn = fmaxf(hn, 0.0f);                        // post-GRU ReLU
    hout[(long)row * 128 + col] = (bf16_t)hn;
  }
}

// ---------------------------------------------------------------------------
extern "C" void kernel_launch(void* const* d_in, const int* in_sizes, int n_in,
                              void* d_out, int out_size, void* d_ws, size_t ws_size,
                              hipStream_t stream) {
  const float* nf  = (const float*)d_in[0];
  const int*   ei  = (const int*)d_in[1];   // [2,E] flat: src then dst
  const int*   et  = (const int*)d_in[2];
  const float* npw = (const float*)d_in[3];
  const float* npb = (const float*)d_in[4];
  const float* eW  = (const float*)d_in[5];
  /* d_in[6] edge_b unused (use_edge_bias=False) */
  const float* wih = (const float*)d_in[7];
  const float* whh = (const float*)d_in[8];
  const float* bih = (const float*)d_in[9];
  const float* bhh = (const float*)d_in[10];
  const float* ow  = (const float*)d_in[11];
  const float* ob  = (const float*)d_in[12];
  float* out = (float*)d_out;

  const int H = 128, T = 8, L = 2;
  const int N = in_sizes[0] / H;            // 100000
  const int E = in_sizes[2];                // 600000
  const int Mtiles = N / 16;                // 6250

  // Workspace carve-up (bf16 buffers), 256B aligned.
  char* ws = (char*)d_ws;
  size_t off = 0;
  auto alloc = [&](size_t bytes) -> char* {
    char* p = ws + off;
    off += (bytes + 255) & ~(size_t)255;
    return p;
  };
  bf16_t* hA    = (bf16_t*)alloc((size_t)N * H * 2);
  bf16_t* hB    = (bf16_t*)alloc((size_t)N * H * 2);
  bf16_t* trans = (bf16_t*)alloc((size_t)T * N * H * 2);   // 204.8 MB
  bf16_t* agg   = (bf16_t*)alloc((size_t)N * H * 2);
  bf16_t* nf16  = (bf16_t*)alloc((size_t)N * H * 2);
  bf16_t* w_np  = (bf16_t*)alloc((size_t)H * H * 2);
  bf16_t* w_e   = (bf16_t*)alloc((size_t)L * T * H * H * 2);
  bf16_t* w_ih  = (bf16_t*)alloc((size_t)L * 3 * H * H * 2);
  bf16_t* w_hh  = (bf16_t*)alloc((size_t)L * 3 * H * H * 2);
  bf16_t* w_o   = (bf16_t*)alloc((size_t)H * H * 2);

  auto cvt = [&](const float* s, bf16_t* d, int n) {
    cvt_f32_bf16<<<(n + 255) / 256, 256, 0, stream>>>(s, d, n);
  };
  cvt(nf,  nf16, N * H);
  cvt(npw, w_np, H * H);
  cvt(eW,  w_e,  L * T * H * H);
  cvt(wih, w_ih, L * 3 * H * H);
  cvt(whh, w_hh, L * 3 * H * H);
  cvt(ow,  w_o,  H * H);

  const dim3 gblk((Mtiles + 7) / 8);        // 782 blocks, 8 waves each

  // h = relu(nf @ Wnp^T + b)
  gemm128<true, bf16_t><<<gblk, 256, 0, stream>>>(
      nf16, w_np, 0, hA, 0, npb, Mtiles);

  bf16_t* hcur = hA;
  bf16_t* hnxt = hB;
  for (int l = 0; l < L; ++l) {
    // trans[t] = h @ W_{l,t}^T  for all t (grid.y = T)
    gemm128<false, bf16_t><<<dim3((Mtiles + 7) / 8, T), 256, 0, stream>>>(
        hcur, w_e + (size_t)l * T * H * H, (long)H * H,
        trans, (long)N * H, nullptr, Mtiles);
    // agg = 0
    zero_u32<<<(N * H / 2 + 255) / 256, 256, 0, stream>>>((u32*)agg, N * H / 2);
    // agg[dst] += trans[type][src]   (packed bf16 atomics, L2-resident)
    scatter_edges<<<(E + 7) / 8, 256, 0, stream>>>(
        trans, ei, ei + E, et, agg, E, N);
    // h' = relu(GRU(agg, h))
    gru_fused<<<(Mtiles * 8 + 7) / 8, 256, 0, stream>>>(
        agg, hcur, w_ih + (size_t)l * 3 * H * H, w_hh + (size_t)l * 3 * H * H,
        bih + l * 3 * H, bhh + l * 3 * H, hnxt, Mtiles);
    bf16_t* t2 = hcur; hcur = hnxt; hnxt = t2;
  }

  // out = h @ out_w^T + out_b   (fp32 output)
  gemm128<false, float><<<gblk, 256, 0, stream>>>(
      hcur, w_o, 0, out, 0, ob, Mtiles);
}
